// RopeAttention_91285234909596
// MI455X (gfx1250) — matrix operand order repaired and checked by reading, not compiled, not run on hardware
//
#include <hip/hip_runtime.h>
#include <hip/hip_bf16.h>
#include <math.h>

// ---------------------------------------------------------------------------
// Types
// ---------------------------------------------------------------------------
typedef __bf16 bf16;
typedef __attribute__((ext_vector_type(16))) __bf16 v16bf;
typedef __attribute__((ext_vector_type(8)))  __bf16 v8bf;
typedef __attribute__((ext_vector_type(8)))  float  v8f;
typedef __attribute__((ext_vector_type(4)))  unsigned int u32x4;
typedef __attribute__((ext_vector_type(8)))  int i32x8;
typedef __attribute__((ext_vector_type(4)))  int i32x4;

#define D_MODEL   2048
#define NUM_HEADS 16
#define HEAD_DIM  128
#define SEQ       2048
#define BATCH     2
#define ROWS      (BATCH * SEQ)          // 4096

#if defined(__HIP_DEVICE_COMPILE__) && __has_builtin(__builtin_amdgcn_tensor_load_to_lds)
#define USE_TDM 1
#else
#define USE_TDM 0
#endif

// ---------------------------------------------------------------------------
// Helpers
// ---------------------------------------------------------------------------
__device__ __forceinline__ v16bf make_frag(const bf16* lo, const bf16* hi) {
    union { v16bf v; v8bf h[2]; } u;
    u.h[0] = *(const v8bf*)lo;
    u.h[1] = *(const v8bf*)hi;
    return u.v;
}
__device__ __forceinline__ v16bf make_frag16(const bf16* p) {
    union { v16bf v; v8bf h[2]; } u;
    u.h[0] = *(const v8bf*)p;
    u.h[1] = *(const v8bf*)(p + 8);
    return u.v;
}
__device__ __forceinline__ v8f wmma_bf16(v16bf a, v16bf b, v8f c) {
    return __builtin_amdgcn_wmma_f32_16x16x32_bf16(
        /*neg_a=*/false, a, /*neg_b=*/false, b,
        /*c_mod=*/(short)0, c, /*reuse_a=*/false, /*reuse_b=*/false);
}

#if USE_TDM
// Tensor Data Mover: 2D bf16 tile, global -> LDS (packed row-major in LDS).
// D# layout per CDNA5 ISA 8.3/8.4: g0 = {count,lds_addr,global_addr,type},
// g1 = {flags/data_size, tensor_dim0/1, tile_dim0/1/2, tensor_dim0_stride}.
__device__ __forceinline__ void tdm_load_2d_bf16(
    unsigned lds_off, const void* gaddr,
    unsigned tile_d0, unsigned tile_d1,        // tile size, elements
    unsigned tensor_d0, unsigned tensor_d1,    // tensor size, elements
    unsigned stride_d0)                        // row stride, elements
{
    unsigned long long ga = (unsigned long long)(uintptr_t)gaddr;
    u32x4 g0;
    g0[0] = 1u;                                          // count=1, user mode
    g0[1] = lds_off;                                     // lds_addr (bytes)
    g0[2] = (unsigned)ga;                                // global_addr[31:0]
    g0[3] = (unsigned)((ga >> 32) & 0x01FFFFFFu) | (2u << 30); // addr[56:32]|type=2
    i32x8 g1;
    g1[0] = (int)(1u << 16);                             // data_size = 2 bytes
    g1[1] = (int)(tensor_d0 << 16);                      // tensor_dim0[15:0]
    g1[2] = (int)((tensor_d0 >> 16) | (tensor_d1 << 16));
    g1[3] = (int)((tensor_d1 >> 16) | (tile_d0 << 16));  // tile_dim0
    g1[4] = (int)tile_d1;                                // tile_dim1, tile_dim2=0
    g1[5] = (int)stride_d0;                              // tensor_dim0_stride lo
    g1[6] = 0;
    g1[7] = 0;
    i32x4 z4 = {0, 0, 0, 0};
#if defined(__clang_major__) && (__clang_major__ >= 23)
    i32x8 z8 = {0, 0, 0, 0, 0, 0, 0, 0};
    __builtin_amdgcn_tensor_load_to_lds(g0, g1, z4, z4, z8, 0);
#else
    __builtin_amdgcn_tensor_load_to_lds(g0, g1, z4, z4, 0);
#endif
}
#define LDS_OFF(p) ((unsigned)(uintptr_t)(p))
#endif

// ---------------------------------------------------------------------------
// Kernel 0a: fp32 -> bf16 (straight)
// ---------------------------------------------------------------------------
__global__ __launch_bounds__(256) void k_cvt_bf16(const float* __restrict__ src,
                                                  bf16* __restrict__ dst) {
    int i = (blockIdx.x * 256 + threadIdx.x) * 8;
    float4 a = *(const float4*)(src + i);
    float4 b = *(const float4*)(src + i + 4);
    v8bf o;
    o[0] = (bf16)a.x; o[1] = (bf16)a.y; o[2] = (bf16)a.z; o[3] = (bf16)a.w;
    o[4] = (bf16)b.x; o[5] = (bf16)b.y; o[6] = (bf16)b.z; o[7] = (bf16)b.w;
    *(v8bf*)(dst + i) = o;
}

// ---------------------------------------------------------------------------
// Kernel 0b: fp32 [K][N] -> bf16 transposed [N][K]  (one-time; removes all
// in-loop scatter transposes from the GEMM kernels)
// ---------------------------------------------------------------------------
__global__ __launch_bounds__(256) void k_cvt_bf16_t(const float* __restrict__ src,
                                                    bf16* __restrict__ dst) {
    int i = (blockIdx.x * 256 + threadIdx.x) * 8;
    int k  = i >> 11;                  // source row
    int n0 = i & (D_MODEL - 1);        // source col start
    float4 a = *(const float4*)(src + i);
    float4 b = *(const float4*)(src + i + 4);
    float v[8] = {a.x, a.y, a.z, a.w, b.x, b.y, b.z, b.w};
#pragma unroll
    for (int e = 0; e < 8; ++e)
        dst[(size_t)(n0 + e) * D_MODEL + k] = (bf16)v[e];
}

// ---------------------------------------------------------------------------
// Kernel 1: fused QKV projection + RoPE.
//   grid = (ROWS/128, NUM_HEADS, 3)   block = 256 (8 waves)
//   Double-buffered LDS tiles; TDM loads issued by wave 0 one step ahead.
//   Q,K stored [B][H][S][Dh];  V stored TRANSPOSED [B][H][Dh][S].
// ---------------------------------------------------------------------------
__global__ __launch_bounds__(256) void k_qkv_rope(
    const bf16* __restrict__ X,        // [ROWS][D_MODEL]
    const bf16* __restrict__ WqT,      // [N][K] transposed
    const bf16* __restrict__ WkT,
    const bf16* __restrict__ WvT,
    bf16* __restrict__ Q, bf16* __restrict__ K, bf16* __restrict__ V,
    const int* __restrict__ pos_ids)   // [ROWS]
{
    const int z = blockIdx.z;
    const bf16* WT = (z == 0) ? WqT : (z == 1) ? WkT : WvT;
    const int h     = blockIdx.y;
    const int row0  = blockIdx.x * 128;
    const int ncol0 = h * HEAD_DIM;

    __shared__ bf16 As[2][128 * 32];   // [row][k]
    __shared__ bf16 Bs[2][128 * 32];   // [n][k]

    const int tid  = threadIdx.x;
    const int wave = tid >> 5;
    const int lane = tid & 31;
    const int m    = lane & 15;
    const int hi   = lane >> 4;
    const int base8 = hi * 8;
    const int hi16  = hi * 16;

    v8f acc[8];
#pragma unroll
    for (int j = 0; j < 8; ++j) acc[j] = (v8f)0.0f;

    const int NK = D_MODEL / 32;       // 64

#if USE_TDM
    if (wave == 0) {
        tdm_load_2d_bf16(LDS_OFF(&As[0][0]), X  + (size_t)row0  * D_MODEL,
                         32, 128, D_MODEL, ROWS,    D_MODEL);
        tdm_load_2d_bf16(LDS_OFF(&Bs[0][0]), WT + (size_t)ncol0 * D_MODEL,
                         32, 128, D_MODEL, D_MODEL, D_MODEL);
    }
    for (int i = 0; i < NK; ++i) {
        const int cur = i & 1;
        if (wave == 0) __builtin_amdgcn_s_wait_tensorcnt(0);
        __syncthreads();                            // publish buf[cur]
        if (wave == 0 && (i + 1) < NK) {
            const int k1 = (i + 1) * 32;
            tdm_load_2d_bf16(LDS_OFF(&As[1 - cur][0]),
                             X  + (size_t)row0  * D_MODEL + k1,
                             32, 128, D_MODEL, ROWS,    D_MODEL);
            tdm_load_2d_bf16(LDS_OFF(&Bs[1 - cur][0]),
                             WT + (size_t)ncol0 * D_MODEL + k1,
                             32, 128, D_MODEL, D_MODEL, D_MODEL);
        }
        const int arow = wave * 16 + m;
        v16bf a = make_frag(&As[cur][arow * 32 + hi16],
                            &As[cur][arow * 32 + hi16 + 8]);
#pragma unroll
        for (int j = 0; j < 8; ++j) {
            int n = j * 16 + m;
            v16bf b = make_frag16(&Bs[cur][n * 32 + hi16]);
            acc[j] = wmma_bf16(a, b, acc[j]);
        }
    }
#else
    for (int i = 0; i < NK; ++i) {
        const int k0 = i * 32;
        __syncthreads();
        {   // stage A: straight 2D copy
            int lin = tid * 16;
            int r   = lin >> 5;
            int c   = lin & 31;
            const bf16* src = X + (size_t)(row0 + r) * D_MODEL + k0 + c;
            *(v8bf*)&As[0][lin]     = *(const v8bf*)src;
            *(v8bf*)&As[0][lin + 8] = *(const v8bf*)(src + 8);
        }
        {   // stage B: straight 2D copy from pre-transposed weights
            int lin = tid * 16;
            int n   = lin >> 5;
            int c   = lin & 31;
            const bf16* src = WT + (size_t)(ncol0 + n) * D_MODEL + k0 + c;
            *(v8bf*)&Bs[0][lin]     = *(const v8bf*)src;
            *(v8bf*)&Bs[0][lin + 8] = *(const v8bf*)(src + 8);
        }
        __syncthreads();
        const int arow = wave * 16 + m;
        v16bf a = make_frag(&As[0][arow * 32 + hi16],
                            &As[0][arow * 32 + hi16 + 8]);
#pragma unroll
        for (int j = 0; j < 8; ++j) {
            int n = j * 16 + m;
            v16bf b = make_frag16(&Bs[0][n * 32 + hi16]);
            acc[j] = wmma_bf16(a, b, acc[j]);
        }
    }
#endif

    // ---- RoPE (q,k): d and d+64 are in acc[j]/acc[j+4] of the same lane ----
    if (z < 2) {
#pragma unroll
        for (int j = 0; j < 4; ++j) {
            int d1 = j * 16 + m;                   // 0..63
            // 10000^(-d/64) = exp(-d * ln(10000)/64)
            float invf = __expf((float)d1 * -0.14391156831212787f);
#pragma unroll
            for (int r = 0; r < 8; ++r) {
                int rr  = row0 + wave * 16 + r + 8 * hi;
                float p = (float)pos_ids[rr];
                float sn, cs;
                __sincosf(p * invf, &sn, &cs);
                float x1 = acc[j][r], x2 = acc[j + 4][r];
                acc[j][r]     = x1 * cs - x2 * sn;
                acc[j + 4][r] = x2 * cs + x1 * sn;
            }
        }
    }

    // ---- store ----
#pragma unroll
    for (int j = 0; j < 8; ++j) {
#pragma unroll
        for (int r = 0; r < 8; ++r) {
            int rr = row0 + wave * 16 + r + 8 * hi;
            int bb = rr >> 11;
            int ss = rr & (SEQ - 1);
            float val = acc[j][r];
            if (z == 2) {
                // V transposed: [B][H][Dh][S]
                size_t addr = ((size_t)(bb * NUM_HEADS + h) * HEAD_DIM
                               + j * 16 + m) * SEQ + ss;
                V[addr] = (bf16)val;
            } else {
                size_t addr = (((size_t)(bb * NUM_HEADS + h) * SEQ) + ss)
                              * HEAD_DIM + j * 16 + m;
                ((z == 0) ? Q : K)[addr] = (bf16)val;
            }
        }
    }
}

// ---------------------------------------------------------------------------
// Kernel 2: flash-style causal attention.
//   grid = (SEQ/128, NUM_HEADS, BATCH)  block = 256 (8 waves)
//   K tile [key][dh] and V tile [dh][key] staged by TDM (V already transposed
//   in global memory, so both are straight 2D tile loads).
// ---------------------------------------------------------------------------
__global__ __launch_bounds__(256) void k_attn(
    const bf16* __restrict__ Q,        // [B][H][S][Dh]
    const bf16* __restrict__ K,        // [B][H][S][Dh]
    const bf16* __restrict__ V,        // [B][H][Dh][S]  (transposed)
    bf16* __restrict__ Aout)           // [B][S][H*Dh]
{
    const int qt = blockIdx.x;
    const int h  = blockIdx.y;
    const int b  = blockIdx.z;
    const int qrow0 = qt * 128;

    const size_t bh = ((size_t)(b * NUM_HEADS + h)) * SEQ * HEAD_DIM;
    const bf16* Qb = Q + bh;
    const bf16* Kb = K + bh;
    const bf16* Vb = V + bh;           // [Dh][S]

    __shared__ bf16 Kt[64 * 128];      // [key][dh]
    __shared__ bf16 Vt[128 * 64];      // [dh][key]
    __shared__ bf16 Pb[8][16 * 64];    // per-wave P scratch

    const int tid  = threadIdx.x;
    const int wave = tid >> 5;
    const int lane = tid & 31;
    const int m    = lane & 15;
    const int hi   = lane >> 4;
    const int base8 = hi * 8;
    const int hi16  = hi * 16;

    v16bf qf[4];
    {
        const int qrow = qrow0 + wave * 16 + m;
        const bf16* qp = Qb + (size_t)qrow * HEAD_DIM;
#pragma unroll
        for (int ks = 0; ks < 4; ++ks)
            qf[ks] = make_frag(qp + ks * 32 + hi16, qp + ks * 32 + hi16 + 8);
    }

    v8f oacc[8];
#pragma unroll
    for (int j = 0; j < 8; ++j) oacc[j] = (v8f)0.0f;
    float mst[8], lst[8];
#pragma unroll
    for (int r = 0; r < 8; ++r) { mst[r] = -3.0e38f; lst[r] = 0.0f; }

    const float scal = 0.08838834764831845f;   // 1/sqrt(128)
    const int nkb = (qt + 1) * 2;

    for (int kb = 0; kb < nkb; ++kb) {
        const int kbase = kb * 64;
        __syncthreads();               // previous tiles fully consumed
#if USE_TDM
        if (wave == 0) {
            // K tile: 64 rows (keys) x 128 dh, row stride HEAD_DIM
            tdm_load_2d_bf16(LDS_OFF(&Kt[0]),
                             Kb + (size_t)kbase * HEAD_DIM,
                             128, 64, HEAD_DIM, SEQ, HEAD_DIM);
            // V tile: 128 rows (dh) x 64 keys, row stride SEQ
            tdm_load_2d_bf16(LDS_OFF(&Vt[0]),
                             Vb + kbase,
                             64, 128, SEQ, HEAD_DIM, SEQ);
            __builtin_amdgcn_s_wait_tensorcnt(0);
        }
        __syncthreads();
#else
#pragma unroll
        for (int i = 0; i < 4; ++i) {
            int lin = tid * 8 + i * 2048;
            *(v8bf*)&Kt[lin] = *(const v8bf*)(Kb + (size_t)kbase * HEAD_DIM + lin);
        }
#pragma unroll
        for (int i = 0; i < 4; ++i) {
            int lin = tid * 8 + i * 2048;
            int dh  = lin >> 6;
            int key = lin & 63;
            *(v8bf*)&Vt[lin] = *(const v8bf*)(Vb + (size_t)dh * SEQ + kbase + key);
        }
        __syncthreads();
#endif

        // --- scores ---
        v8f sc[4];
#pragma unroll
        for (int nt = 0; nt < 4; ++nt) {
            sc[nt] = (v8f)0.0f;
            int key = nt * 16 + m;
#pragma unroll
            for (int ks = 0; ks < 4; ++ks) {
                v16bf bfrag = make_frag16(&Kt[key * 128 + ks * 32 + hi16]);
                sc[nt] = wmma_bf16(qf[ks], bfrag, sc[nt]);
            }
        }
        // scale + causal mask
#pragma unroll
        for (int nt = 0; nt < 4; ++nt) {
            int kg = kbase + nt * 16 + m;
#pragma unroll
            for (int r = 0; r < 8; ++r) {
                int qg = qrow0 + wave * 16 + r + 8 * hi;
                float v = sc[nt][r] * scal;
                if (kg > qg) v += -1000000000.0f;
                sc[nt][r] = v;
            }
        }
        // --- online softmax ---
        float mx[8];
#pragma unroll
        for (int r = 0; r < 8; ++r) {
            float v = sc[0][r];
            v = fmaxf(v, sc[1][r]); v = fmaxf(v, sc[2][r]); v = fmaxf(v, sc[3][r]);
            v = fmaxf(v, __shfl_xor(v, 1));
            v = fmaxf(v, __shfl_xor(v, 2));
            v = fmaxf(v, __shfl_xor(v, 4));
            v = fmaxf(v, __shfl_xor(v, 8));
            mx[r] = v;
        }
        float alpha[8], rsum[8];
#pragma unroll
        for (int r = 0; r < 8; ++r) {
            float mnew = fmaxf(mst[r], mx[r]);
            alpha[r] = __expf(mst[r] - mnew);
            mst[r] = mnew;
            rsum[r] = 0.0f;
        }
#pragma unroll
        for (int nt = 0; nt < 4; ++nt) {
#pragma unroll
            for (int r = 0; r < 8; ++r) {
                float p = __expf(sc[nt][r] - mst[r]);
                rsum[r] += p;
                Pb[wave][(r + 8 * hi) * 64 + nt * 16 + m] = (bf16)p;
            }
        }
#pragma unroll
        for (int r = 0; r < 8; ++r) {
            float v = rsum[r];
            v += __shfl_xor(v, 1);
            v += __shfl_xor(v, 2);
            v += __shfl_xor(v, 4);
            v += __shfl_xor(v, 8);
            lst[r] = lst[r] * alpha[r] + v;
        }
#pragma unroll
        for (int j = 0; j < 8; ++j)
#pragma unroll
            for (int r = 0; r < 8; ++r) oacc[j][r] *= alpha[r];

        // --- P @ V ---
#pragma unroll
        for (int jt = 0; jt < 8; ++jt) {
            int dh = jt * 16 + m;
#pragma unroll
            for (int ks = 0; ks < 2; ++ks) {
                v16bf a = make_frag(&Pb[wave][m * 64 + ks * 32 + hi16],
                                    &Pb[wave][m * 64 + ks * 32 + hi16 + 8]);
                v16bf bb = make_frag16(&Vt[dh * 64 + ks * 32 + hi16]);
                oacc[jt] = wmma_bf16(a, bb, oacc[jt]);
            }
        }
    }

    // --- epilogue ---
#pragma unroll
    for (int jt = 0; jt < 8; ++jt) {
#pragma unroll
        for (int r = 0; r < 8; ++r) {
            int s = qrow0 + wave * 16 + r + 8 * hi;
            float o = oacc[jt][r] / lst[r];
            size_t addr = ((size_t)(b * SEQ + s)) * D_MODEL + h * HEAD_DIM
                        + jt * 16 + m;
            Aout[addr] = (bf16)o;
        }
    }
}

// ---------------------------------------------------------------------------
// Kernel 3: output projection (bf16 x bf16^T -> fp32), TDM double-buffered.
//   grid = (ROWS/128, D_MODEL/128)  block = 256
// ---------------------------------------------------------------------------
__global__ __launch_bounds__(256) void k_oproj(
    const bf16* __restrict__ A,        // [ROWS][D_MODEL]
    const bf16* __restrict__ WT,       // [N][K] transposed
    float* __restrict__ Out)           // [ROWS][D_MODEL]
{
    const int row0  = blockIdx.x * 128;
    const int ncol0 = blockIdx.y * 128;

    __shared__ bf16 As[2][128 * 32];
    __shared__ bf16 Bs[2][128 * 32];

    const int tid  = threadIdx.x;
    const int wave = tid >> 5;
    const int lane = tid & 31;
    const int m    = lane & 15;
    const int hi   = lane >> 4;
    const int base8 = hi * 8;
    const int hi16  = hi * 16;

    v8f acc[8];
#pragma unroll
    for (int j = 0; j < 8; ++j) acc[j] = (v8f)0.0f;

    const int NK = D_MODEL / 32;

#if USE_TDM
    if (wave == 0) {
        tdm_load_2d_bf16(LDS_OFF(&As[0][0]), A  + (size_t)row0  * D_MODEL,
                         32, 128, D_MODEL, ROWS,    D_MODEL);
        tdm_load_2d_bf16(LDS_OFF(&Bs[0][0]), WT + (size_t)ncol0 * D_MODEL,
                         32, 128, D_MODEL, D_MODEL, D_MODEL);
    }
    for (int i = 0; i < NK; ++i) {
        const int cur = i & 1;
        if (wave == 0) __builtin_amdgcn_s_wait_tensorcnt(0);
        __syncthreads();
        if (wave == 0 && (i + 1) < NK) {
            const int k1 = (i + 1) * 32;
            tdm_load_2d_bf16(LDS_OFF(&As[1 - cur][0]),
                             A  + (size_t)row0  * D_MODEL + k1,
                             32, 128, D_MODEL, ROWS,    D_MODEL);
            tdm_load_2d_bf16(LDS_OFF(&Bs[1 - cur][0]),
                             WT + (size_t)ncol0 * D_MODEL + k1,
                             32, 128, D_MODEL, D_MODEL, D_MODEL);
        }
        const int arow = wave * 16 + m;
        v16bf a = make_frag(&As[cur][arow * 32 + hi16],
                            &As[cur][arow * 32 + hi16 + 8]);
#pragma unroll
        for (int j = 0; j < 8; ++j) {
            int n = j * 16 + m;
            v16bf b = make_frag16(&Bs[cur][n * 32 + hi16]);
            acc[j] = wmma_bf16(a, b, acc[j]);
        }
    }
#else
    for (int i = 0; i < NK; ++i) {
        const int k0 = i * 32;
        __syncthreads();
        {
            int lin = tid * 16;
            int r   = lin >> 5;
            int c   = lin & 31;
            const bf16* src = A + (size_t)(row0 + r) * D_MODEL + k0 + c;
            *(v8bf*)&As[0][lin]     = *(const v8bf*)src;
            *(v8bf*)&As[0][lin + 8] = *(const v8bf*)(src + 8);
        }
        {
            int lin = tid * 16;
            int n   = lin >> 5;
            int c   = lin & 31;
            const bf16* src = WT + (size_t)(ncol0 + n) * D_MODEL + k0 + c;
            *(v8bf*)&Bs[0][lin]     = *(const v8bf*)src;
            *(v8bf*)&Bs[0][lin + 8] = *(const v8bf*)(src + 8);
        }
        __syncthreads();
        const int arow = wave * 16 + m;
        v16bf a = make_frag(&As[0][arow * 32 + hi16],
                            &As[0][arow * 32 + hi16 + 8]);
#pragma unroll
        for (int j = 0; j < 8; ++j) {
            int n = j * 16 + m;
            v16bf b = make_frag16(&Bs[0][n * 32 + hi16]);
            acc[j] = wmma_bf16(a, b, acc[j]);
        }
    }
#endif

#pragma unroll
    for (int j = 0; j < 8; ++j) {
#pragma unroll
        for (int r = 0; r < 8; ++r) {
            int rr = row0 + wave * 16 + r + 8 * hi;
            Out[(size_t)rr * D_MODEL + ncol0 + j * 16 + m] = acc[j][r];
        }
    }
}

// ---------------------------------------------------------------------------
// Launcher
// ---------------------------------------------------------------------------
extern "C" void kernel_launch(void* const* d_in, const int* in_sizes, int n_in,
                              void* d_out, int out_size, void* d_ws, size_t ws_size,
                              hipStream_t stream) {
    const float* Xf  = (const float*)d_in[0];
    // d_in[1] = attention_mask (causal; reproduced in-kernel)
    const int*   pos = (const int*)d_in[2];
    const float* Wqf = (const float*)d_in[3];
    const float* Wkf = (const float*)d_in[4];
    const float* Wvf = (const float*)d_in[5];
    const float* Wof = (const float*)d_in[6];

    const size_t MB = 1024 * 1024;
    char* ws = (char*)d_ws;
    bf16* Xb  = (bf16*)(ws + 0 * MB);     // 16 MiB : [4096][2048]
    bf16* WqT = (bf16*)(ws + 16 * MB);    //  8 MiB : transposed
    bf16* WkT = (bf16*)(ws + 24 * MB);
    bf16* WvT = (bf16*)(ws + 32 * MB);
    bf16* WoT = (bf16*)(ws + 40 * MB);
    bf16* Qb  = (bf16*)(ws + 48 * MB);    // 16 MiB : [B][H][S][Dh]
    bf16* Kb  = (bf16*)(ws + 64 * MB);
    bf16* Vb  = (bf16*)(ws + 80 * MB);    // 16 MiB : [B][H][Dh][S]
    bf16* Ab  = (bf16*)(ws + 96 * MB);    // 16 MiB : [B][S][H*Dh]

    const int nX = ROWS * D_MODEL;
    const int nW = D_MODEL * D_MODEL;
    k_cvt_bf16  <<<nX / 2048, 256, 0, stream>>>(Xf,  Xb);
    k_cvt_bf16_t<<<nW / 2048, 256, 0, stream>>>(Wqf, WqT);
    k_cvt_bf16_t<<<nW / 2048, 256, 0, stream>>>(Wkf, WkT);
    k_cvt_bf16_t<<<nW / 2048, 256, 0, stream>>>(Wvf, WvT);
    k_cvt_bf16_t<<<nW / 2048, 256, 0, stream>>>(Wof, WoT);

    dim3 gA(ROWS / 128, NUM_HEADS, 3);
    k_qkv_rope<<<gA, 256, 0, stream>>>(Xb, WqT, WkT, WvT, Qb, Kb, Vb, pos);

    dim3 gB(SEQ / 128, NUM_HEADS, BATCH);
    k_attn<<<gB, 256, 0, stream>>>(Qb, Kb, Vb, Ab);

    dim3 gC(ROWS / 128, D_MODEL / 128);
    k_oproj<<<gC, 256, 0, stream>>>(Ab, WoT, (float*)d_out);
}
